// DeepseekV4Indexer_78752520339582
// MI455X (gfx1250) — matrix operand order repaired and checked by reading, not compiled, not run on hardware
//
#include <hip/hip_runtime.h>

// ---------------------------------------------------------------------------
// DeepSeek-V4 indexer on MI455X (gfx1250): bf16 WMMA for all GEMMs,
// async global->LDS double-buffered staging (ASYNCcnt path).
// T=2048, HID=7168, QR=1536, H=64, D=128, TOPK=256, R=4, C=512
// ---------------------------------------------------------------------------

typedef __attribute__((ext_vector_type(16))) __bf16 bf16x16;
typedef __attribute__((ext_vector_type(8)))  __bf16 bf16x8;
typedef __attribute__((ext_vector_type(8)))  float  f32x8;

#define T_TOK   2048
#define HID     7168
#define QRD     1536
#define NH      64
#define DD      128
#define TOPK    256
#define RR      4
#define CC      512
#define NKV     512      // 4*D columns of W_fused
#define NQ      8192     // H*D

__device__ __forceinline__ __bf16 f2bf(float f) {
  unsigned int u = __builtin_bit_cast(unsigned int, f);
  u += 0x7fffu + ((u >> 16) & 1u);           // round-to-nearest-even
  unsigned short s = (unsigned short)(u >> 16);
  return __builtin_bit_cast(__bf16, s);
}

// gfx1250 async copy: global -> LDS, 16 bytes per lane, tracked by ASYNCcnt.
__device__ __forceinline__ void async_b128_to_lds(const __bf16* gptr,
                                                  __bf16* lptr) {
  unsigned lds_off = (unsigned)(unsigned long long)lptr;  // low 32b = LDS offset
  unsigned long long ga = (unsigned long long)gptr;
  asm volatile("global_load_async_to_lds_b128 %0, %1, off"
               :: "v"(lds_off), "v"(ga)
               : "memory");
}
__device__ __forceinline__ void wait_asynccnt0() {
  asm volatile("s_wait_asynccnt 0x0" ::: "memory");
}

// ---------------------------------------------------------------------------
// fp32 -> bf16 conversion
// ---------------------------------------------------------------------------
__global__ void cvt_bf16_kernel(const float* __restrict__ src,
                                __bf16* __restrict__ dst, int n) {
  int i = blockIdx.x * blockDim.x + threadIdx.x;
  if (i < n) dst[i] = f2bf(src[i]);
}

// ---------------------------------------------------------------------------
// Generic GEMM:  Out[M x N] (f32) = alpha * A[M x K](bf16) @ W[N x K](bf16)^T
// Block: 256 threads = 8 waves. Tile: M=128 (16 rows/wave), N=64, Kstep=32.
// B tile async-copied to LDS, double-buffered; A read direct (L2-hot).
// ---------------------------------------------------------------------------
__global__ __launch_bounds__(256)
void gemm_bf16_nt(const __bf16* __restrict__ A, const __bf16* __restrict__ W,
                  float* __restrict__ Out, int M, int N, int K, float alpha) {
  __shared__ __bf16 sB[2][64 * 32] __attribute__((aligned(32)));
  const int tid  = threadIdx.x;
  const int wave = tid >> 5;
  const int lane = tid & 31;
  const int ln   = lane & 15;
  const int h    = lane >> 4;
  const int mbase = blockIdx.y * 128;
  const int nbase = blockIdx.x * 64;
  const int arow  = mbase + wave * 16 + ln;
  // this thread's staging slot: element e of the 64x32 tile
  const int se  = tid * 8;
  const int sr  = se >> 5, sk = se & 31;

  f32x8 acc[4];
#pragma unroll
  for (int j = 0; j < 4; ++j) acc[j] = 0.0f;

  // prologue: kick off tile 0
  async_b128_to_lds(W + (size_t)(nbase + sr) * K + sk, &sB[0][se]);

  const int nit = K / 32;
  for (int it = 0; it < nit; ++it) {
    const int kt  = it * 32;
    const int cur = it & 1;
    wait_asynccnt0();        // this wave's writes for tile `it` landed
    __syncthreads();         // all waves' writes landed; prev reads done
    if (it + 1 < nit)        // stream tile it+1 while computing tile it
      async_b128_to_lds(W + (size_t)(nbase + sr) * K + kt + 32 + sk,
                        &sB[cur ^ 1][se]);

    // A fragment: lane ln holds row arow, K runs at kt+h*16 .. kt+h*16+15
    bf16x16 afrag;
    {
      const __bf16* ap = A + (size_t)arow * K + kt + h * 16;
      bf16x8 lo = *(const bf16x8*)(ap);
      bf16x8 hi = *(const bf16x8*)(ap + 8);
#pragma unroll
      for (int i = 0; i < 8; ++i) { afrag[i] = lo[i]; afrag[8 + i] = hi[i]; }
    }
    if (kt + 32 < K)
      __builtin_prefetch(A + (size_t)arow * K + kt + 32, 0, 3); // global_prefetch_b8

#pragma unroll
    for (int j = 0; j < 4; ++j) {
      // B fragment: lane = K (halves), VGPR pair = consecutive K of column n
      bf16x16 bfrag = *(const bf16x16*)(&sB[cur][(j * 16 + ln) * 32 + h * 16]);
      acc[j] = __builtin_amdgcn_wmma_f32_16x16x32_bf16(
          false, afrag, false, bfrag, (short)0, acc[j], false, false);
    }
  }

#pragma unroll
  for (int j = 0; j < 4; ++j)
#pragma unroll
    for (int p = 0; p < 8; ++p) {
      int m = mbase + wave * 16 + p + h * 8;
      int n = nbase + j * 16 + ln;
      Out[(size_t)m * N + n] = acc[j][p] * alpha;
    }
}

// ---------------------------------------------------------------------------
// q GEMM with fused RoPE + D^-0.5 scale, bf16 output.
// A = qr_bf16 (T x 1536), W = Wq_bf16 (8192 x 1536).
// N-tile = 128 = one head => RoPE partner lives in accumulator j^4.
// Output layout: Q[(t*64 + head)*128 + d]  (head = blockIdx.x)
// ---------------------------------------------------------------------------
__global__ __launch_bounds__(256)
void gemm_rope_q(const __bf16* __restrict__ A, const __bf16* __restrict__ W,
                 const int* __restrict__ positions,
                 const float* __restrict__ csc, __bf16* __restrict__ Qout) {
  const int K = QRD;
  __shared__ __bf16 sB[2][128 * 32] __attribute__((aligned(32)));
  const int tid  = threadIdx.x;
  const int wave = tid >> 5;
  const int lane = tid & 31;
  const int ln   = lane & 15;
  const int h    = lane >> 4;
  const int mbase = blockIdx.y * 128;      // tokens
  const int nbase = blockIdx.x * 128;      // head * 128
  const int arow  = mbase + wave * 16 + ln;
  const float alpha = 0.08838834764831845f;  // D^-0.5
  // staging: 128x32 tile = 4096 elems, 2 x b128 per thread
  const int se0 = tid * 8, se1 = tid * 8 + 2048;
  const int sr0 = se0 >> 5, sk0 = se0 & 31;
  const int sr1 = se1 >> 5, sk1 = se1 & 31;

  f32x8 acc[8];
#pragma unroll
  for (int j = 0; j < 8; ++j) acc[j] = 0.0f;

  async_b128_to_lds(W + (size_t)(nbase + sr0) * K + sk0, &sB[0][se0]);
  async_b128_to_lds(W + (size_t)(nbase + sr1) * K + sk1, &sB[0][se1]);

  const int nit = K / 32;
  for (int it = 0; it < nit; ++it) {
    const int kt  = it * 32;
    const int cur = it & 1;
    wait_asynccnt0();
    __syncthreads();
    if (it + 1 < nit) {
      async_b128_to_lds(W + (size_t)(nbase + sr0) * K + kt + 32 + sk0,
                        &sB[cur ^ 1][se0]);
      async_b128_to_lds(W + (size_t)(nbase + sr1) * K + kt + 32 + sk1,
                        &sB[cur ^ 1][se1]);
    }

    bf16x16 afrag;
    {
      const __bf16* ap = A + (size_t)arow * K + kt + h * 16;
      bf16x8 lo = *(const bf16x8*)(ap);
      bf16x8 hi = *(const bf16x8*)(ap + 8);
#pragma unroll
      for (int i = 0; i < 8; ++i) { afrag[i] = lo[i]; afrag[8 + i] = hi[i]; }
    }
    if (kt + 32 < K)
      __builtin_prefetch(A + (size_t)arow * K + kt + 32, 0, 3);

#pragma unroll
    for (int j = 0; j < 8; ++j) {
      bf16x16 bfrag = *(const bf16x16*)(&sB[cur][(j * 16 + ln) * 32 + h * 16]);
      acc[j] = __builtin_amdgcn_wmma_f32_16x16x32_bf16(
          false, afrag, false, bfrag, (short)0, acc[j], false, false);
    }
  }

  // fused RoPE epilogue: x1' = x1*cos - x2*sin ; x2' = x2*cos + x1*sin
  const int m0 = mbase + wave * 16;
  const int head = blockIdx.x;
#pragma unroll
  for (int p = 0; p < 8; ++p) {
    int m = m0 + p + h * 8;                   // token
    int ps = positions[m];
    const float* cb = csc + (size_t)ps * 128;
#pragma unroll
    for (int j = 0; j < 8; ++j) {
      int nloc = j * 16 + ln;
      int jj = nloc & 63;
      float cs = cb[jj], sn = cb[64 + jj];
      float self  = acc[j][p]     * alpha;
      float other = acc[j ^ 4][p] * alpha;
      float res = (j < 4) ? (self * cs - other * sn)
                          : (self * cs + other * sn);
      Qout[((size_t)m * NH + head) * DD + nloc] = f2bf(res);
    }
  }
}

// ---------------------------------------------------------------------------
// Window-softmax compress + RMS norm + RoPE -> k (bf16, C x D)
// kv_score row: [0,128)=kv0 [128,256)=kv1 [256,384)=sc0 [384,512)=sc1
// ---------------------------------------------------------------------------
__global__ __launch_bounds__(128)
void compress_rope_k(const float* __restrict__ kvs, const float* __restrict__ ape,
                     const float* __restrict__ rmsw, const float* __restrict__ csc,
                     __bf16* __restrict__ Kout) {
  const int c = blockIdx.x;
  const int d = threadIdx.x;
  float g[8], v[8];
#pragma unroll
  for (int r = 0; r < RR; ++r) {
    if (c == 0) { g[r] = -1e30f; v[r] = 0.0f; }
    else {
      int t = (c - 1) * RR + r;
      g[r] = kvs[(size_t)t * NKV + 256 + d] + ape[r * DD + d];
      v[r] = kvs[(size_t)t * NKV + d];
    }
    int t2 = c * RR + r;
    g[4 + r] = kvs[(size_t)t2 * NKV + 384 + d] + ape[(RR + r) * DD + d];
    v[4 + r] = kvs[(size_t)t2 * NKV + 128 + d];
  }
  float mx = g[0];
#pragma unroll
  for (int i = 1; i < 8; ++i) mx = fmaxf(mx, g[i]);
  float se = 0.0f, sv = 0.0f;
#pragma unroll
  for (int i = 0; i < 8; ++i) {
    float e = __expf(g[i] - mx);
    se += e; sv += e * v[i];
  }
  float comp = sv / se;

  __shared__ float s[128];
  s[d] = comp * comp;
  __syncthreads();
  for (int off = 64; off > 0; off >>= 1) {
    if (d < off) s[d] += s[d + off];
    __syncthreads();
  }
  float rs = rsqrtf(s[0] * (1.0f / 128.0f) + 1e-6f);
  __syncthreads();
  float nrm = comp * rs * rmsw[d];
  s[d] = nrm;
  __syncthreads();
  int j = d & 63;
  float cs = csc[(size_t)c * 128 + j], sn = csc[(size_t)c * 128 + 64 + j];
  float res = (d < 64) ? (nrm * cs - s[d + 64] * sn)
                       : (nrm * cs + s[d - 64] * sn);
  Kout[(size_t)c * DD + d] = f2bf(res);
}

// ---------------------------------------------------------------------------
// qk GEMM with fused relu + head reduction:
// scores[t,c] = sum_h wts[t,h] * relu(q[t,h,:] . k[c,:])
// A = Q (T*H x 128), B = k (512 x 128). M-tile 128 = 2 tokens x 64 heads.
// ---------------------------------------------------------------------------
__global__ __launch_bounds__(256)
void qk_scores(const __bf16* __restrict__ Q, const __bf16* __restrict__ Kc,
               const float* __restrict__ wts, float* __restrict__ scores) {
  __shared__ float sacc[2][64];
  const int tid  = threadIdx.x;
  const int wave = tid >> 5;
  const int lane = tid & 31;
  const int ln   = lane & 15;
  const int h    = lane >> 4;
  const int mbase = blockIdx.y * 128;   // rows of (T*H)
  const int nbase = blockIdx.x * 64;    // chunk index c
  const int arow  = mbase + wave * 16 + ln;

  f32x8 acc[4];
#pragma unroll
  for (int j = 0; j < 4; ++j) acc[j] = 0.0f;

#pragma unroll
  for (int kt = 0; kt < DD; kt += 32) {
    bf16x16 afrag;
    {
      const __bf16* ap = Q + (size_t)arow * DD + kt + h * 16;
      bf16x8 lo = *(const bf16x8*)(ap);
      bf16x8 hi = *(const bf16x8*)(ap + 8);
#pragma unroll
      for (int i = 0; i < 8; ++i) { afrag[i] = lo[i]; afrag[8 + i] = hi[i]; }
    }
#pragma unroll
    for (int j = 0; j < 4; ++j) {
      bf16x16 bfrag = *(const bf16x16*)(Kc + (size_t)(nbase + j * 16 + ln) * DD
                                        + kt + h * 16);
      acc[j] = __builtin_amdgcn_wmma_f32_16x16x32_bf16(
          false, afrag, false, bfrag, (short)0, acc[j], false, false);
    }
  }

  // epilogue: contrib = wts[t,h] * relu(val); reduce over the 64 heads
  const int m0 = mbase + wave * 16;
  const int t  = m0 >> 6;                 // token (wave lies inside one token)
  float wv[8];
#pragma unroll
  for (int p = 0; p < 8; ++p) {
    int m = m0 + p + h * 8;
    wv[p] = wts[(size_t)t * NH + (m & 63)];
  }
  if (tid < 128) sacc[tid >> 6][tid & 63] = 0.0f;
  __syncthreads();
#pragma unroll
  for (int j = 0; j < 4; ++j) {
    float s = 0.0f;
#pragma unroll
    for (int p = 0; p < 8; ++p) s += wv[p] * fmaxf(acc[j][p], 0.0f);
    s += __shfl_xor(s, 16, 32);           // combine rows p and p+8
    if (lane < 16) atomicAdd(&sacc[wave >> 2][j * 16 + ln], s); // ds_add_f32
  }
  __syncthreads();
  if (tid < 128) {
    int tl = tid >> 6, n = tid & 63;
    scores[(size_t)((mbase >> 6) + tl) * CC + nbase + n] = sacc[tl][n];
  }
}

// ---------------------------------------------------------------------------
// Per-token masked top-256 of 512 via bitonic sort (value desc, index asc).
// ---------------------------------------------------------------------------
__global__ __launch_bounds__(256)
void topk_kernel(const float* __restrict__ scores,
                 const int* __restrict__ positions, int* __restrict__ out) {
  const int t = blockIdx.x;
  const int tid = threadIdx.x;
  __shared__ float sv[CC];
  __shared__ int   si[CC];
  const int ncomp = (positions[t] + 1) >> 2;   // (pos+1)/R
  for (int i = tid; i < CC; i += 256) {
    sv[i] = (i < ncomp) ? scores[(size_t)t * CC + i] : -1e30f;
    si[i] = i;
  }
  __syncthreads();
  for (int k = 2; k <= CC; k <<= 1)
    for (int jj = k >> 1; jj > 0; jj >>= 1) {
      for (int i = tid; i < CC; i += 256) {
        int ix = i ^ jj;
        if (ix > i) {
          float va = sv[i], vb = sv[ix];
          int   ia = si[i], ib = si[ix];
          bool b_first = (vb > va) || (vb == va && ib < ia);
          bool a_first = (va > vb) || (va == vb && ia < ib);
          bool asc = ((i & k) == 0);
          if (asc ? b_first : a_first) {
            sv[i] = vb; sv[ix] = va; si[i] = ib; si[ix] = ia;
          }
        }
      }
      __syncthreads();
    }
  int valid = min(ncomp, TOPK);
  if (tid < TOPK)
    out[(size_t)t * TOPK + tid] = (tid < valid) ? si[tid] : -1;
}

// ---------------------------------------------------------------------------
// Workspace layout (bytes, each region 256B aligned)
// ---------------------------------------------------------------------------
#define OFF_HSB   ((size_t)0)                       // T*HID bf16     = 29360128
#define OFF_QRB   (OFF_HSB + (size_t)T_TOK*HID*2)   // T*QRD bf16     =  6291456
#define OFF_WFB   (OFF_QRB + (size_t)T_TOK*QRD*2)   // NKV*HID bf16   =  7340032
#define OFF_WQB   (OFF_WFB + (size_t)NKV*HID*2)     // NQ*QRD bf16    = 25165824
#define OFF_WPB   (OFF_WQB + (size_t)NQ*QRD*2)      // NH*HID bf16    =   917504
#define OFF_KVS   (OFF_WPB + (size_t)NH*HID*2)      // T*NKV f32      =  4194304
#define OFF_WTS   (OFF_KVS + (size_t)T_TOK*NKV*4)   // T*NH f32       =   524288
#define OFF_KBF   (OFF_WTS + (size_t)T_TOK*NH*4)    // CC*DD bf16     =   131072
#define OFF_QBF   (OFF_KBF + (size_t)CC*DD*2)       // T*NH*DD bf16   = 33554432
#define OFF_SCO   (OFF_QBF + (size_t)T_TOK*NH*DD*2) // T*CC f32       =  4194304

extern "C" void kernel_launch(void* const* d_in, const int* in_sizes, int n_in,
                              void* d_out, int out_size, void* d_ws, size_t ws_size,
                              hipStream_t stream) {
  (void)in_sizes; (void)n_in; (void)out_size; (void)ws_size;
  const float* hs   = (const float*)d_in[0];
  const float* qr   = (const float*)d_in[1];
  const int*   pos  = (const int*)  d_in[2];
  const float* Wf   = (const float*)d_in[3];
  const float* Wq   = (const float*)d_in[4];
  const float* Wp   = (const float*)d_in[5];
  const float* ape  = (const float*)d_in[6];
  const float* rmsw = (const float*)d_in[7];
  const float* csc  = (const float*)d_in[8];

  char* ws = (char*)d_ws;
  __bf16* hsb = (__bf16*)(ws + OFF_HSB);
  __bf16* qrb = (__bf16*)(ws + OFF_QRB);
  __bf16* wfb = (__bf16*)(ws + OFF_WFB);
  __bf16* wqb = (__bf16*)(ws + OFF_WQB);
  __bf16* wpb = (__bf16*)(ws + OFF_WPB);
  float*  kvs = (float*) (ws + OFF_KVS);
  float*  wts = (float*) (ws + OFF_WTS);
  __bf16* kbf = (__bf16*)(ws + OFF_KBF);
  __bf16* qbf = (__bf16*)(ws + OFF_QBF);
  float*  sco = (float*) (ws + OFF_SCO);

  auto cvt = [&](const float* s, __bf16* d, int n) {
    cvt_bf16_kernel<<<(n + 255) / 256, 256, 0, stream>>>(s, d, n);
  };
  cvt(hs, hsb, T_TOK * HID);
  cvt(qr, qrb, T_TOK * QRD);
  cvt(Wf, wfb, NKV * HID);
  cvt(Wq, wqb, NQ * QRD);
  cvt(Wp, wpb, NH * HID);

  // kv_score = hs @ W_fused^T        (2048 x 512, K=7168)
  gemm_bf16_nt<<<dim3(NKV / 64, T_TOK / 128), 256, 0, stream>>>(
      hsb, wfb, kvs, T_TOK, NKV, HID, 1.0f);
  // wts = hs @ Wproj^T * H^-0.5      (2048 x 64,  K=7168)
  gemm_bf16_nt<<<dim3(NH / 64, T_TOK / 128), 256, 0, stream>>>(
      hsb, wpb, wts, T_TOK, NH, HID, 0.125f);
  // q = rope(qr @ Wq^T) * D^-0.5     (2048 x 8192, K=1536), bf16 out
  gemm_rope_q<<<dim3(NQ / 128, T_TOK / 128), 256, 0, stream>>>(
      qrb, wqb, pos, csc, qbf);
  // k = rope(rmsnorm(compress(kv_score)))
  compress_rope_k<<<CC, 128, 0, stream>>>(kvs, ape, rmsw, csc, kbf);
  // scores = sum_h wts * relu(q . k)
  qk_scores<<<dim3(CC / 64, (T_TOK * NH) / 128), 256, 0, stream>>>(
      qbf, kbf, wts, sco);
  // masked top-256 indices
  topk_kernel<<<T_TOK, 256, 0, stream>>>(sco, pos, (int*)d_out);
}
